// KANLayer_28363964023654
// MI455X (gfx1250) — matrix operand order repaired and checked by reading, not compiled, not run on hardware
//
#include <hip/hip_runtime.h>
#include <hip/hip_bf16.h>

typedef __attribute__((ext_vector_type(16))) _Float16 v16h_t;
typedef __attribute__((ext_vector_type(8)))  _Float16 v8h_t;
typedef __attribute__((ext_vector_type(8)))  float    v8f_t;

#define IN_DIM   2048
#define OUT_DIM  2048
#define BATCH    4096
#define GRID_G   8
#define K2       (2 * IN_DIM)   // fused K dimension = 4096
#define LN_EPS   1e-5f

// LDS row stride in halfs (96B: keeps every 8-half segment 16B-aligned)
#define LROW 48

// ---------------------------------------------------------------------------
// CDNA5 async global->LDS copy (16B per lane), tracked by ASYNCcnt.
// VDST = per-lane LDS byte address, VADDR = per-lane 64-bit global address.
// ---------------------------------------------------------------------------
__device__ __forceinline__ void async_b128(uint32_t lds_addr, uint64_t gaddr) {
    asm volatile("global_load_async_to_lds_b128 %0, %1, off"
                 :: "v"(lds_addr), "v"(gaddr)
                 : "memory");
}
#define WAIT_ASYNC(n) asm volatile("s_wait_asynccnt " #n ::: "memory")

__device__ __forceinline__ uint32_t lds_u32(const void* p) {
    // generic shared address: addr[31:0] == LDS byte offset (ISA 10.2 aperture map)
    return (uint32_t)(uintptr_t)p;
}

// ---------------------------------------------------------------------------
// Kernel 1: Wc[o][0..IN) = (f16) scale_base[o][i]
//           Wc[o][IN..2IN) = (f16) sum_g spline_weight[o][i][g]
// ---------------------------------------------------------------------------
__global__ __launch_bounds__(256) void prep_w_kernel(
    const float* __restrict__ sw,   // (OUT, IN, G)
    const float* __restrict__ sb,   // (OUT, IN)
    _Float16* __restrict__ Wc)      // (OUT, 2*IN) f16
{
    const size_t gid = (size_t)blockIdx.x * 256 + threadIdx.x;   // < OUT*IN
    const float4* p = (const float4*)(sw + gid * GRID_G);
    float4 a = p[0], b = p[1];
    float s = a.x + a.y + a.z + a.w + b.x + b.y + b.z + b.w;

    const int o = (int)(gid >> 11);
    const int i = (int)(gid & (IN_DIM - 1));
    _Float16* row = Wc + (size_t)o * K2;
    row[i]          = (_Float16)sb[gid];
    row[IN_DIM + i] = (_Float16)s;
}

// ---------------------------------------------------------------------------
// Kernel 2: LayerNorm -> xn ; S = sum_g exp(-(xn-g)^2 * beta)
//           Ac[b][0..IN) = (f16) xn ; Ac[b][IN..2IN) = (f16) S
// ---------------------------------------------------------------------------
__global__ __launch_bounds__(256) void prep_a_kernel(
    const float* __restrict__ x,
    const float* __restrict__ lnw,
    const float* __restrict__ lnb,
    const float* __restrict__ rbf_beta,
    const float* __restrict__ grid,
    _Float16* __restrict__ Ac)
{
    __shared__ float s1[256];
    __shared__ float s2[256];
    const int b   = blockIdx.x;
    const int tid = threadIdx.x;

    const float4* xr = (const float4*)(x + (size_t)b * IN_DIM + tid * 8);
    float4 v0 = xr[0], v1 = xr[1];
    float xv[8] = {v0.x, v0.y, v0.z, v0.w, v1.x, v1.y, v1.z, v1.w};

    float sum = 0.f, sq = 0.f;
#pragma unroll
    for (int j = 0; j < 8; ++j) { sum += xv[j]; sq += xv[j] * xv[j]; }
    s1[tid] = sum; s2[tid] = sq;
    __syncthreads();
#pragma unroll
    for (int s = 128; s > 0; s >>= 1) {
        if (tid < s) { s1[tid] += s1[tid + s]; s2[tid] += s2[tid + s]; }
        __syncthreads();
    }
    const float mu   = s1[0] * (1.0f / IN_DIM);
    const float var  = s2[0] * (1.0f / IN_DIM) - mu * mu;
    const float rstd = rsqrtf(var + LN_EPS);
    const float beta = fminf(fmaxf(rbf_beta[0], 0.5f), 6.0f);

    float g[GRID_G];
#pragma unroll
    for (int gg = 0; gg < GRID_G; ++gg) g[gg] = grid[gg];

    const float4* wr = (const float4*)(lnw + tid * 8);
    const float4* br = (const float4*)(lnb + tid * 8);
    float4 w0 = wr[0], w1 = wr[1], b0 = br[0], b1 = br[1];
    float wv[8] = {w0.x, w0.y, w0.z, w0.w, w1.x, w1.y, w1.z, w1.w};
    float bv[8] = {b0.x, b0.y, b0.z, b0.w, b1.x, b1.y, b1.z, b1.w};

    _Float16* rowX = Ac + (size_t)b * K2 + tid * 8;
    _Float16* rowS = rowX + IN_DIM;
#pragma unroll
    for (int j = 0; j < 8; ++j) {
        float xn = (xv[j] - mu) * rstd * wv[j] + bv[j];
        float S = 0.f;
#pragma unroll
        for (int gg = 0; gg < GRID_G; ++gg) {
            float d = xn - g[gg];
            S += __expf(-d * d * beta);
        }
        rowX[j] = (_Float16)xn;
        rowS[j] = (_Float16)S;
    }
}

// ---------------------------------------------------------------------------
// Kernel 3: C(MxN) = A(MxK2) * W(NxK2)^T + bias, f16 WMMA, f32 accum.
// Block tile 256x128, K-step 32, 8 waves (4 M-groups x 2 N-groups),
// wave tile 64x64 = 4x4 WMMA 16x16x32 -> 16 WMMAs per 16 ds_load_b128.
// Double-buffered LDS filled with GLOBAL_LOAD_ASYNC_TO_LDS_B128 (ASYNCcnt),
// prefetching tile k+1 while computing tile k. Async-copy global addresses
// are persistent registers advanced by 64B/step (6 v_add_nc_u64 per K-step);
// LDS destinations are loop-invariant per buffer.
// ---------------------------------------------------------------------------
__global__ __launch_bounds__(256) void gemm_wmma_kernel(
    const _Float16* __restrict__ A,    // (BATCH, K2)
    const _Float16* __restrict__ W,    // (OUT, K2)
    const float* __restrict__ bias,    // (OUT)
    float* __restrict__ C)             // (BATCH, OUT)
{
    __shared__ _Float16 As[2][256][LROW];   // 48 KB
    __shared__ _Float16 Bs[2][128][LROW];   // 24 KB

    const int tid  = threadIdx.x;
    const int lane = tid & 31;
    const int wid  = tid >> 5;
    const int wm   = wid >> 1;    // 0..3 : M group of 64 rows
    const int wn   = wid & 1;     // 0..1 : N group of 64 cols
    const int m16  = lane & 15;
    const int ksel = lane >> 4;
    const int blockM = blockIdx.y * 256;
    const int blockN = blockIdx.x * 128;

    // per-thread copy coordinates (row, 8-half segment)
    const int arow0 = tid >> 2, aseg = tid & 3;   // + u*64 rows

    // Loop-invariant LDS destinations (12 u32 constants, 6 per buffer)
    uint32_t aldst[2][4], bldst[2][2];
#pragma unroll
    for (int bb = 0; bb < 2; ++bb) {
#pragma unroll
        for (int u = 0; u < 4; ++u)
            aldst[bb][u] = lds_u32(&As[bb][arow0 + u * 64][aseg * 8]);
#pragma unroll
        for (int u = 0; u < 2; ++u)
            bldst[bb][u] = lds_u32(&Bs[bb][arow0 + u * 64][aseg * 8]);
    }

    // Persistent global source pointers: advance 64B per K-step
    uint64_t ag[4], bg[2];
#pragma unroll
    for (int u = 0; u < 4; ++u)
        ag[u] = (uint64_t)(uintptr_t)(A + (size_t)(blockM + arow0 + u * 64) * K2 + aseg * 8);
#pragma unroll
    for (int u = 0; u < 2; ++u)
        bg[u] = (uint64_t)(uintptr_t)(W + (size_t)(blockN + arow0 + u * 64) * K2 + aseg * 8);

    v8f_t acc[4][4];
#pragma unroll
    for (int mi = 0; mi < 4; ++mi)
#pragma unroll
        for (int ni = 0; ni < 4; ++ni) acc[mi][ni] = {};

    // ---- preload tile k0=0 into buffer 0 -------------------------------
#pragma unroll
    for (int u = 0; u < 4; ++u) { async_b128(aldst[0][u], ag[u]); ag[u] += 64; }
#pragma unroll
    for (int u = 0; u < 2; ++u) { async_b128(bldst[0][u], bg[u]); bg[u] += 64; }

#pragma unroll 2
    for (int k0 = 0; k0 < K2; k0 += 32) {
        const int buf = (k0 >> 5) & 1;

        if (k0 + 32 < K2) {
            // prefetch next K-tile into the other buffer
#pragma unroll
            for (int u = 0; u < 4; ++u) { async_b128(aldst[buf ^ 1][u], ag[u]); ag[u] += 64; }
#pragma unroll
            for (int u = 0; u < 2; ++u) { async_b128(bldst[buf ^ 1][u], bg[u]); bg[u] += 64; }
            WAIT_ASYNC(6);   // the 6 copies for THIS buffer are complete
        } else {
            WAIT_ASYNC(0);
        }
        __syncthreads();     // all waves' copies for `buf` visible

        v16h_t af[4], bf[4];
#pragma unroll
        for (int mi = 0; mi < 4; ++mi) {
            int r = wm * 64 + mi * 16 + m16;
            v8h_t lo = *(const v8h_t*)&As[buf][r][ksel * 8];
            v8h_t hi = *(const v8h_t*)&As[buf][r][ksel * 8 + 16];
            af[mi] = __builtin_shufflevector(lo, hi, 0,1,2,3,4,5,6,7,8,9,10,11,12,13,14,15);
        }
#pragma unroll
        for (int ni = 0; ni < 4; ++ni) {
            int r = wn * 64 + ni * 16 + m16;
            v8h_t lo = *(const v8h_t*)&Bs[buf][r][ksel * 8];
            v8h_t hi = *(const v8h_t*)&Bs[buf][r][ksel * 8 + 16];
            bf[ni] = __builtin_shufflevector(lo, hi, 0,1,2,3,4,5,6,7,8,9,10,11,12,13,14,15);
        }

#pragma unroll
        for (int mi = 0; mi < 4; ++mi)
#pragma unroll
            for (int ni = 0; ni < 4; ++ni)
                acc[mi][ni] = __builtin_amdgcn_wmma_f32_16x16x32_f16(
                    false, af[mi], false, bf[ni], (short)0, acc[mi][ni], false, false);

        __syncthreads();     // everyone done reading `buf` before it is refilled
    }

    // Epilogue: VGPR r holds M = r + (lane>>4)*8, N = lane&15 (ISA C/D layout)
#pragma unroll
    for (int mi = 0; mi < 4; ++mi) {
#pragma unroll
        for (int ni = 0; ni < 4; ++ni) {
            int col = blockN + wn * 64 + ni * 16 + m16;
            int rowBase = blockM + wm * 64 + mi * 16 + ksel * 8;
            float bv = bias[col];
#pragma unroll
            for (int r2 = 0; r2 < 8; ++r2)
                C[(size_t)(rowBase + r2) * OUT_DIM + col] = acc[mi][ni][r2] + bv;
        }
    }
}

extern "C" void kernel_launch(void* const* d_in, const int* in_sizes, int n_in,
                              void* d_out, int out_size, void* d_ws, size_t ws_size,
                              hipStream_t stream) {
    const float* x    = (const float*)d_in[0];
    const float* lnw  = (const float*)d_in[1];
    const float* lnb  = (const float*)d_in[2];
    const float* sw   = (const float*)d_in[3];
    const float* sb   = (const float*)d_in[4];
    const float* bias = (const float*)d_in[5];
    const float* beta = (const float*)d_in[6];
    const float* grid = (const float*)d_in[7];
    float* out = (float*)d_out;

    _Float16* Ac = (_Float16*)d_ws;                                   // 32 MB
    _Float16* Wc = (_Float16*)((char*)d_ws + (size_t)BATCH * K2 * 2); // 16 MB

    prep_w_kernel<<<(OUT_DIM * IN_DIM) / 256, 256, 0, stream>>>(sw, sb, Wc);
    prep_a_kernel<<<BATCH, 256, 0, stream>>>(x, lnw, lnb, beta, grid, Ac);

    dim3 g(OUT_DIM / 128, BATCH / 256);
    gemm_wmma_kernel<<<g, 256, 0, stream>>>(Ac, Wc, bias, out);
}
